// DenoisingGate_16037407883993
// MI455X (gfx1250) — matrix operand order repaired and checked by reading, not compiled, not run on hardware
//
#include <hip/hip_runtime.h>

typedef __attribute__((ext_vector_type(16))) _Float16 v16h;
typedef __attribute__((ext_vector_type(8)))  _Float16 v8h;
typedef __attribute__((ext_vector_type(8)))  float    v8f;

#define HID 128
#define EPB 128   // edges per block (8 waves x 16)
#define TPB 256

// ---- prep: convert node features f32 -> f16 ----
__global__ void cvt_x_f16(const float* __restrict__ x, _Float16* __restrict__ xf, int n) {
  int i = blockIdx.x * blockDim.x + threadIdx.x;
  int stride = gridDim.x * blockDim.x;
  for (; i < n; i += stride) xf[i] = (_Float16)x[i];
}

// ---- prep: swizzle W1 (256x128 used rows) and W2 (128x64) into WMMA B-fragment order.
// B fragment (32x16, f16): lane l holds column n=l&15; half h=l>>4, element j carries
// K=8h+j for j<8 and K=16+8h+(j-8) for j>=8.
__global__ void build_wfrags(const float* __restrict__ W1, const float* __restrict__ W2,
                             _Float16* __restrict__ w1f, _Float16* __restrict__ w2f) {
  int gid = blockIdx.x * blockDim.x + threadIdx.x;
  const int N1 = 8 * 8 * 512;            // 8 ksteps x 8 ntiles x (32 lanes * 16 halfs)
  if (gid < N1) {
    int frag = gid >> 9, within = gid & 511;
    int lane = within >> 4, j = within & 15;
    int s = frag >> 3, t = frag & 7;
    int K = s * 32 + (lane >> 4) * 8 + (j & 7) + (j >> 3) * 16;
    int N = t * 16 + (lane & 15);
    w1f[gid] = (_Float16)W1[K * HID + N];
  } else if (gid < N1 + 4 * 4 * 512) {   // W2: 4 ksteps x 4 ntiles
    int idx = gid - N1;
    int frag = idx >> 9, within = idx & 511;
    int lane = within >> 4, j = within & 15;
    int s = frag >> 2, t = frag & 3;
    int K = s * 32 + (lane >> 4) * 8 + (j & 7) + (j >> 3) * 16;
    int N = t * 16 + (lane & 15);
    w2f[idx] = (_Float16)W2[K * 64 + N];
  }
}

static __device__ __forceinline__ v16h loadB(const _Float16* base, int frag, int lane) {
  return *(const v16h*)(base + ((long long)frag * 32 + lane) * 16);
}

static __device__ __forceinline__ v16h loadA16(const _Float16* row) {
  union { v16h v; v8h h[2]; } a;
  a.h[0] = *(const v8h*)(row);        // K chunk j=0..7
  a.h[1] = *(const v8h*)(row + 16);   // K chunk j=8..15
  return a.v;
}

// ---- main fused kernel: gather -> GEMM1(WMMA) -> +w*W1_last+b1 -> LN -> ReLU
//      -> GEMM2(WMMA) -> +b2 -> ReLU -> dot(W3)+b3 -> sigmoid ----
__launch_bounds__(TPB, 1)
__global__ void gate_kernel(const _Float16* __restrict__ xf,
                            const long long* __restrict__ eidx,
                            const float* __restrict__ ew,
                            const float* __restrict__ W1,
                            const float* __restrict__ b1,
                            const float* __restrict__ gam,
                            const float* __restrict__ bet,
                            const float* __restrict__ b2,
                            const float* __restrict__ W3,
                            const float* __restrict__ b3,
                            const _Float16* __restrict__ w1f,
                            const _Float16* __restrict__ w2f,
                            float* __restrict__ out, int E) {
  // 64KB region: phase1 = W1 fragments (f16), phase2 = h f32 [128][128], phase4 = h2 f32 [128][64]
  __shared__ float sBufA[EPB * HID];
  __shared__ _Float16 sH16[EPB * HID];      // 32KB: post-LN activations, f16 row-major
  __shared__ _Float16 sW2f[4 * 4 * 512];    // 16KB: W2 fragments
  __shared__ float sB1[HID], sGam[HID], sBet[HID], sW1L[HID];
  __shared__ float sB2[64], sW3[64];
  __shared__ float sEw[EPB];
  __shared__ int sRow[EPB], sCol[EPB];
  __shared__ float sP0[TPB], sP1[TPB];      // cross-thread partial sums
  __shared__ float sB3;

  const int tid = threadIdx.x;
  const long long ebase = (long long)blockIdx.x * EPB;

  // ---- stage weights / params / edge metadata ----
  _Float16* sW1f = (_Float16*)sBufA;
  {
    const uint4* s1 = (const uint4*)w1f;    // 64KB
    uint4* d1 = (uint4*)sW1f;
    for (int i = tid; i < 4096; i += TPB) d1[i] = s1[i];
    const uint4* s2 = (const uint4*)w2f;    // 16KB
    uint4* d2 = (uint4*)sW2f;
    for (int i = tid; i < 1024; i += TPB) d2[i] = s2[i];
  }
  if (tid < HID) {
    sB1[tid] = b1[tid]; sGam[tid] = gam[tid]; sBet[tid] = bet[tid];
    sW1L[tid] = W1[256 * HID + tid];        // edge-weight column of W1
  }
  if (tid < 64) { sB2[tid] = b2[tid]; sW3[tid] = W3[tid]; }
  if (tid == 0) sB3 = b3[0];
  if (tid < EPB) {
    long long e = ebase + tid;
    long long ec = e < E ? e : (long long)(E - 1);
    sEw[tid] = ew[ec];
    sRow[tid] = (int)eidx[ec];
    sCol[tid] = (int)eidx[(long long)E + ec];
  }
  __syncthreads();

  const int wave = tid >> 5, lane = tid & 31;
  const int hi = lane >> 4, l15 = lane & 15;
  const int eloc = wave * 16 + l15;         // edge row this lane feeds for A fragments
  const long long nodeR = sRow[eloc];       // hoisted: loop-invariant LDS reads
  const long long nodeC = sCol[eloc];
  const _Float16* xrR = xf + nodeR * HID + hi * 8;
  const _Float16* xrC = xf + nodeC * HID + hi * 8;

  // ---- GEMM1: [16 edges x 256] @ [256 x 128], f16 WMMA, f32 accum,
  //      software-pipelined A (global, L2 hits) and B (LDS) fragments ----
  v8f c[8] = {};
  v16h a_cur = loadA16(xrR);                // s = 0
#pragma unroll
  for (int s = 0; s < 8; ++s) {
    v16h a_next = a_cur;
    if (s < 7) {
      int sn = s + 1;
      const _Float16* base = (sn < 4) ? xrR : xrC;
      a_next = loadA16(base + (sn & 3) * 32);
    }
    v16h b_cur = loadB(sW1f, s * 8, lane);
#pragma unroll
    for (int t = 0; t < 8; ++t) {
      v16h b_next = b_cur;
      if (t < 7) b_next = loadB(sW1f, s * 8 + t + 1, lane);
      c[t] = __builtin_amdgcn_wmma_f32_16x16x32_f16(false, a_cur, false, b_cur,
                                                    (short)0, c[t], false, false);
      b_cur = b_next;
    }
    a_cur = a_next;
  }
  __syncthreads();                          // all done reading sW1f

  // ---- emit h = C + w_e * W1[256,:] + b1 into LDS (f32), reusing W1 buffer ----
  float* hbuf = sBufA;
#pragma unroll
  for (int t = 0; t < 8; ++t) {
    int n = t * 16 + l15;
#pragma unroll
    for (int v = 0; v < 8; ++v) {
      int er = wave * 16 + hi * 8 + v;      // C layout: VGPR v, lane-half hi -> row
      hbuf[er * HID + n] = c[t][v] + sEw[er] * sW1L[n] + sB1[n];
    }
  }
  __syncthreads();

  // ---- LayerNorm + ReLU: 2 threads per edge (64 features each) ----
  {
    int e = tid >> 1, half = tid & 1;
    const float* hr = hbuf + e * HID + half * 64;
    float s0 = 0.f, s1 = 0.f;
#pragma unroll 8
    for (int i = 0; i < 64; ++i) { float v = hr[i]; s0 += v; s1 += v * v; }
    sP0[tid] = s0; sP1[tid] = s1;
    __syncthreads();
    float t0 = sP0[e * 2] + sP0[e * 2 + 1];
    float t1 = sP1[e * 2] + sP1[e * 2 + 1];
    float mu = t0 * (1.0f / HID);
    float var = t1 * (1.0f / HID) - mu * mu;
    float rstd = rsqrtf(var + 1e-5f);
    _Float16* hh = sH16 + e * HID + half * 64;
    const float* g = sGam + half * 64;
    const float* bb = sBet + half * 64;
#pragma unroll 8
    for (int i = 0; i < 64; ++i) {
      float v = (hr[i] - mu) * rstd * g[i] + bb[i];
      hh[i] = (_Float16)fmaxf(v, 0.f);
    }
  }
  __syncthreads();

  // ---- GEMM2: [16 x 128] @ [128 x 64], f16 WMMA, pipelined B ----
  v8f c2[4] = {};
  const _Float16* h16row = sH16 + (wave * 16 + l15) * HID + hi * 8;
  v16h a2_cur = loadA16(h16row);            // s = 0
#pragma unroll
  for (int s = 0; s < 4; ++s) {
    v16h a2_next = a2_cur;
    if (s < 3) a2_next = loadA16(h16row + (s + 1) * 32);
    v16h b_cur = loadB(sW2f, s * 4, lane);
#pragma unroll
    for (int t = 0; t < 4; ++t) {
      v16h b_next = b_cur;
      if (t < 3) b_next = loadB(sW2f, s * 4 + t + 1, lane);
      c2[t] = __builtin_amdgcn_wmma_f32_16x16x32_f16(false, a2_cur, false, b_cur,
                                                     (short)0, c2[t], false, false);
      b_cur = b_next;
    }
    a2_cur = a2_next;
  }

  // ---- h2 = ReLU(C2 + b2) into LDS (reuse bufA; h f32 dead since last barrier) ----
  float* h2 = sBufA;
#pragma unroll
  for (int t = 0; t < 4; ++t) {
    int n = t * 16 + l15;
#pragma unroll
    for (int v = 0; v < 8; ++v) {
      int er = wave * 16 + hi * 8 + v;
      h2[er * 64 + n] = fmaxf(c2[t][v] + sB2[n], 0.f);
    }
  }
  __syncthreads();

  // ---- layer 3: sigmoid(h2 . W3 + b3), 2 threads per edge ----
  {
    int e = tid >> 1, half = tid & 1;
    const float* hr = h2 + e * 64 + half * 32;
    const float* wr = sW3 + half * 32;
    float acc = 0.f;
#pragma unroll 8
    for (int i = 0; i < 32; ++i) acc += hr[i] * wr[i];
    sP0[tid] = acc;
    __syncthreads();
    if (half == 0) {
      float tot = sP0[e * 2] + sP0[e * 2 + 1] + sB3;
      float sc = 1.0f / (1.0f + __expf(-tot));
      long long eg = ebase + e;
      if (eg < E) out[eg] = sc;
    }
  }
}

extern "C" void kernel_launch(void* const* d_in, const int* in_sizes, int n_in,
                              void* d_out, int out_size, void* d_ws, size_t ws_size,
                              hipStream_t stream) {
  const float*     x   = (const float*)d_in[0];
  const long long* ei  = (const long long*)d_in[1];
  const float*     ew  = (const float*)d_in[2];
  const float*     W1  = (const float*)d_in[3];
  const float*     b1  = (const float*)d_in[4];
  const float*     gam = (const float*)d_in[5];
  const float*     bet = (const float*)d_in[6];
  const float*     W2  = (const float*)d_in[7];
  const float*     b2  = (const float*)d_in[8];
  const float*     W3  = (const float*)d_in[9];
  const float*     b3  = (const float*)d_in[10];

  int nX = in_sizes[0];            // n_nodes * 128
  int E  = in_sizes[2];            // n_edges

  char* ws = (char*)d_ws;
  size_t xfBytes = (size_t)nX * sizeof(_Float16);
  size_t off = (xfBytes + 255) & ~(size_t)255;
  size_t need = off + 65536 + 16384;
  if (ws_size < need) return;      // scratch too small; cannot run

  _Float16* xf  = (_Float16*)ws;
  _Float16* w1f = (_Float16*)(ws + off);
  _Float16* w2f = (_Float16*)(ws + off + 65536);

  cvt_x_f16<<<(nX + 255) / 256, 256, 0, stream>>>(x, xf, nX);
  build_wfrags<<<(8 * 8 * 512 + 4 * 4 * 512 + 255) / 256, 256, 0, stream>>>(W1, W2, w1f, w2f);

  int nBlocks = (E + EPB - 1) / EPB;
  gate_kernel<<<nBlocks, TPB, 0, stream>>>(xf, ei, ew, W1, b1, gam, bet, b2, W3, b3,
                                           w1f, w2f, (float*)d_out, E);

  // pass-through edge_index after the E gate scores, clamped to d_out capacity
  size_t availBytes = (out_size > E) ? (size_t)(out_size - E) * sizeof(float) : 0;
  size_t wantBytes  = (size_t)2 * (size_t)E * sizeof(long long);
  size_t cpBytes    = wantBytes < availBytes ? wantBytes : availBytes;
  if (cpBytes)
    hipMemcpyAsync((char*)d_out + (size_t)E * sizeof(float), d_in[1], cpBytes,
                   hipMemcpyDeviceToDevice, stream);
}
